// CrossModalBlock_14070312862083
// MI455X (gfx1250) — matrix operand order repaired and checked by reading, not compiled, not run on hardware
//
#include <hip/hip_runtime.h>
#include <math.h>

// CrossModalBlock for MI455X (gfx1250): all GEMMs via v_wmma_f32_16x16x32_bf16
// (bf16 operands, fp32 accumulate). Raw attention scores go straight to d_out
// (they are outputs w1/w2); softmax is applied on-the-fly while staging the
// probs tile for probs @ V, so the 134MB score tensors are read exactly once.
//
// Round-2 changes: 128x128 block tiles (8 waves, wave tile 64x32, 8 WMMA per
// K-step) for 2x arithmetic intensity, and all staging addresses precomputed
// outside the K-loop (pointer += 32 per step) to kill per-iteration v_mul_u64.

#define APITCH 40   // LDS row pitch in halves (16B-aligned rows, conflict pad)
#define BPITCH 40

typedef __attribute__((ext_vector_type(16))) __bf16 v16bf;
typedef __attribute__((ext_vector_type(8)))  float  v8f;

union ABFrag { v16bf v; uint4 u[2]; };

// pack two floats into two RNE-rounded bf16s in one dword
__device__ __forceinline__ unsigned pk_bf(float a, float b) {
  unsigned ua = __float_as_uint(a), ub = __float_as_uint(b);
  ua = (ua + 0x7FFFu + ((ua >> 16) & 1u)) >> 16;
  ub = (ub + 0x7FFFu + ((ub >> 16) & 1u)) >> 16;
  return ua | (ub << 16);
}

// ---------------------------------------------------------------------------
// Generic bf16-WMMA GEMM: C = act((A @ W^T + bias) * alpha) [+ resid]
// A: M x K (lda), W: N x K (ldb), C: M x N (ldc). Optional batching over
// blockIdx.z with (b,h) split strides (used for Q*K^T scores).
// Block: 256 threads = 8 waves; block tile 128x128; wave tile 64x32; K-step 32.
// ---------------------------------------------------------------------------
__global__ __launch_bounds__(256)
void gemm_bf16_kernel(const float* __restrict__ A, const float* __restrict__ Bw,
                      const float* __restrict__ bias, const float* __restrict__ resid,
                      float* __restrict__ C,
                      int Kd, int lda, int ldb, int ldc,
                      long sAb, long sAh, long sBb, long sBh, long sCz,
                      int Hdiv, float alpha, int act)
{
  __shared__ unsigned short As[128 * APITCH];
  __shared__ unsigned short Bs[128 * BPITCH];

  const int m0 = blockIdx.y * 128;
  const int n0 = blockIdx.x * 128;
  const int z  = blockIdx.z;
  const int bb = z / Hdiv, hh = z - bb * Hdiv;
  A  += (size_t)bb * sAb + (size_t)hh * sAh;
  Bw += (size_t)bb * sBb + (size_t)hh * sBh;
  C  += (size_t)z * sCz;

  const int tid  = threadIdx.x;
  const int lane = tid & 31, wv = tid >> 5;
  const int wm   = (wv >> 2) * 64;     // 0 or 64
  const int wn   = (wv & 3) * 32;      // 0, 32, 64, 96
  const int l15  = lane & 15;
  const int hi   = (lane & 16) ? 1 : 0;

  // staging: thread covers rows srow + 32*i (i<4), K-cols scol..scol+3
  const int srow = tid >> 3;
  const int scol = (tid & 7) << 2;
  const float* ag = A  + (size_t)(m0 + srow) * lda + scol;
  const float* bg = Bw + (size_t)(n0 + srow) * ldb + scol;
  unsigned short* asw = &As[srow * APITCH + scol];
  unsigned short* bsw = &Bs[srow * BPITCH + scol];

  // fragment LDS read pointers (loop-invariant)
  const unsigned short* apA[4];
#pragma unroll
  for (int i = 0; i < 4; ++i) apA[i] = &As[(wm + i * 16 + l15) * APITCH + hi * 8];
  const unsigned short* apB[2];
#pragma unroll
  for (int j = 0; j < 2; ++j) apB[j] = &Bs[(wn + j * 16 + l15) * BPITCH + hi * 8];

  v8f acc[4][2] = {};

  for (int k0 = 0; k0 < Kd; k0 += 32) {
#pragma unroll
    for (int i = 0; i < 4; ++i) {
      const float4 fa = *(const float4*)(ag + (size_t)(i * 32) * lda);
      uint2 pa; pa.x = pk_bf(fa.x, fa.y); pa.y = pk_bf(fa.z, fa.w);
      *(uint2*)(asw + i * 32 * APITCH) = pa;
      const float4 fb = *(const float4*)(bg + (size_t)(i * 32) * ldb);
      uint2 pb; pb.x = pk_bf(fb.x, fb.y); pb.y = pk_bf(fb.z, fb.w);
      *(uint2*)(bsw + i * 32 * BPITCH) = pb;
    }
    ag += 32; bg += 32;
    __syncthreads();

    ABFrag afr[4], bfr[2];
#pragma unroll
    for (int i = 0; i < 4; ++i) {  // A: lane<16 K{0-7,16-23}; lane>=16 K{8-15,24-31}
      afr[i].u[0] = *(const uint4*)(apA[i]);
      afr[i].u[1] = *(const uint4*)(apA[i] + 16);
    }
#pragma unroll
    for (int j = 0; j < 2; ++j) {  // B: lane<16 K{0-7,16-23}; lane>=16 K{8-15,24-31} (col = lane&15)
      bfr[j].u[0] = *(const uint4*)(apB[j]);
      bfr[j].u[1] = *(const uint4*)(apB[j] + 16);
    }
#pragma unroll
    for (int i = 0; i < 4; ++i)
#pragma unroll
      for (int j = 0; j < 2; ++j)
        acc[i][j] = __builtin_amdgcn_wmma_f32_16x16x32_bf16(
            false, afr[i].v, false, bfr[j].v, (short)0, acc[i][j], false, false);
    __syncthreads();
  }

  // epilogue: VGPR r -> row (lane<16: M=r ; lane>=16: M=8+r), col = lane&15
#pragma unroll
  for (int i = 0; i < 4; ++i) {
    int rbase = m0 + wm + i * 16 + hi * 8;
#pragma unroll
    for (int j = 0; j < 2; ++j) {
      int col = n0 + wn + j * 16 + l15;
      float bv = bias ? bias[col] : 0.f;
#pragma unroll
      for (int r = 0; r < 8; ++r) {
        int row = rbase + r;
        float v = (acc[i][j][r] + bv) * alpha;
        if (act == 1) v = 0.5f * v * (1.f + erff(v * 0.70710678118654752f));
        if (resid) v += resid[(size_t)row * ldc + col];
        C[(size_t)row * ldc + col] = v;
      }
    }
  }
}

// ---------------------------------------------------------------------------
// Softmax row stats over raw scores: per row (length L) -> max, sum(exp(s-max))
// One wave per row, 8 rows per 256-thread block.
// ---------------------------------------------------------------------------
__global__ __launch_bounds__(256)
void softmax_stats_kernel(const float* __restrict__ S, float* __restrict__ rmax,
                          float* __restrict__ rsum, int L)
{
  const int row  = blockIdx.x * 8 + (threadIdx.x >> 5);
  const int lane = threadIdx.x & 31;
  const float* sr = S + (size_t)row * L;
  float m = -3.402823e38f;
  for (int j = lane; j < L; j += 32) m = fmaxf(m, sr[j]);
#pragma unroll
  for (int o = 16; o; o >>= 1) m = fmaxf(m, __shfl_xor(m, o, 32));
  float s = 0.f;
  for (int j = lane; j < L; j += 32) s += expf(sr[j] - m);
#pragma unroll
  for (int o = 16; o; o >>= 1) s += __shfl_xor(s, o, 32);
  if (lane == 0) { rmax[row] = m; rsum[row] = s; }
}

// ---------------------------------------------------------------------------
// ctx = softmax(S) @ V, softmax applied while staging the P tile to LDS.
// Per (b,h): M=L rows(i), N=64 cols(d), K=L (j). Block tile 128x64 (full head),
// 256 threads = 8 waves, wave tile 32x32.
// ---------------------------------------------------------------------------
__global__ __launch_bounds__(256)
void attn_ctx_kernel(const float* __restrict__ S, const float* __restrict__ rmax,
                     const float* __restrict__ rsum, const float* __restrict__ V,
                     float* __restrict__ Cx, int L, int E, int Hh)
{
  __shared__ unsigned short Ps[128 * APITCH];  // probs tile: 128 i x 32 j
  __shared__ unsigned short Vt[64 * BPITCH];   // V^T tile:    64 d x 32 j

  const int i0 = blockIdx.x * 128;
  const int z  = blockIdx.y;
  const int bb = z / Hh, hh = z - bb * Hh;
  const float* Sg = S + (size_t)z * L * L;
  const float* mg = rmax + (size_t)z * L;
  const float* sg = rsum + (size_t)z * L;
  const float* Vg = V  + (size_t)bb * L * E + hh * 64;
  float*       Cg = Cx + (size_t)bb * L * E + hh * 64;

  const int tid  = threadIdx.x;
  const int lane = tid & 31, wv = tid >> 5;
  const int wm   = (wv >> 1) * 32;   // 0..96 (4 row-groups)
  const int wn   = (wv & 1) * 32;    // 0, 32
  const int l15  = lane & 15;
  const int hi   = (lane & 16) ? 1 : 0;

  // P staging: thread covers rows srow + 32*i (i<4), j-cols scol..scol+3.
  // softmax stats for those 4 rows are loop-invariant -> hoist.
  const int srow = tid >> 3, scol = (tid & 7) << 2;
  const float* sp = Sg + (size_t)(i0 + srow) * L + scol;
  unsigned short* psw = &Ps[srow * APITCH + scol];
  float mx[4], inv[4];
#pragma unroll
  for (int i = 0; i < 4; ++i) {
    mx[i]  = mg[i0 + srow + 32 * i];
    inv[i] = 1.f / sg[i0 + srow + 32 * i];
  }
  // V^T staging: thread covers j-rows jr + 16*i (i<2), d-cols d4..d4+3
  const int jr = tid >> 4, d4 = (tid & 15) << 2;
  const float* vp = Vg + (size_t)jr * E + d4;

  const unsigned short* apA[2];
#pragma unroll
  for (int i = 0; i < 2; ++i) apA[i] = &Ps[(wm + i * 16 + l15) * APITCH + hi * 8];
  const unsigned short* apB[2];
#pragma unroll
  for (int j = 0; j < 2; ++j) apB[j] = &Vt[(wn + j * 16 + l15) * BPITCH + hi * 8];

  v8f acc[2][2] = {};

  for (int j0 = 0; j0 < L; j0 += 32) {
#pragma unroll
    for (int i = 0; i < 4; ++i) {
      const float4 s = *(const float4*)(sp + (size_t)(i * 32) * L);
      uint2 p;
      p.x = pk_bf(expf(s.x - mx[i]) * inv[i], expf(s.y - mx[i]) * inv[i]);
      p.y = pk_bf(expf(s.z - mx[i]) * inv[i], expf(s.w - mx[i]) * inv[i]);
      *(uint2*)(psw + i * 32 * APITCH) = p;
    }
#pragma unroll
    for (int i = 0; i < 2; ++i) {
      const float4 f = *(const float4*)(vp + (size_t)(i * 16) * E);
      int jj = jr + i * 16;
      Vt[(d4 + 0) * BPITCH + jj] = (unsigned short)(pk_bf(f.x, 0.f) & 0xFFFFu);
      Vt[(d4 + 1) * BPITCH + jj] = (unsigned short)(pk_bf(f.y, 0.f) & 0xFFFFu);
      Vt[(d4 + 2) * BPITCH + jj] = (unsigned short)(pk_bf(f.z, 0.f) & 0xFFFFu);
      Vt[(d4 + 3) * BPITCH + jj] = (unsigned short)(pk_bf(f.w, 0.f) & 0xFFFFu);
    }
    sp += 32; vp += (size_t)32 * E;
    __syncthreads();

    ABFrag afr[2], bfr[2];
#pragma unroll
    for (int i = 0; i < 2; ++i) {
      afr[i].u[0] = *(const uint4*)(apA[i]);
      afr[i].u[1] = *(const uint4*)(apA[i] + 16);
    }
#pragma unroll
    for (int j = 0; j < 2; ++j) {
      bfr[j].u[0] = *(const uint4*)(apB[j]);
      bfr[j].u[1] = *(const uint4*)(apB[j] + 16);
    }
#pragma unroll
    for (int i = 0; i < 2; ++i)
#pragma unroll
      for (int j = 0; j < 2; ++j)
        acc[i][j] = __builtin_amdgcn_wmma_f32_16x16x32_bf16(
            false, afr[i].v, false, bfr[j].v, (short)0, acc[i][j], false, false);
    __syncthreads();
  }

#pragma unroll
  for (int i = 0; i < 2; ++i) {
    int rbase = i0 + wm + i * 16 + hi * 8;
#pragma unroll
    for (int j = 0; j < 2; ++j) {
      int col = wn + j * 16 + l15;
#pragma unroll
      for (int r = 0; r < 8; ++r)
        Cg[(size_t)(rbase + r) * E + col] = acc[i][j][r];
    }
  }
}

// ---------------------------------------------------------------------------
// Row LayerNorm over 512 features, one 256-thread block per row.
// ---------------------------------------------------------------------------
__global__ __launch_bounds__(256)
void ln_kernel(const float* __restrict__ x, const float* __restrict__ g,
               const float* __restrict__ bta, float* __restrict__ y)
{
  const int row = blockIdx.x;
  const int t = threadIdx.x;
  const float* xr = x + (size_t)row * 512;
  float a0 = xr[t], a1 = xr[t + 256];
  __shared__ float sm[256], sq[256];
  sm[t] = a0 + a1;
  sq[t] = a0 * a0 + a1 * a1;
  __syncthreads();
  for (int o = 128; o; o >>= 1) {
    if (t < o) { sm[t] += sm[t + o]; sq[t] += sq[t + o]; }
    __syncthreads();
  }
  float mean = sm[0] * (1.f / 512.f);
  float var  = sq[0] * (1.f / 512.f) - mean * mean;
  float inv  = rsqrtf(var + 1e-5f);
  float* yr = y + (size_t)row * 512;
  yr[t]       = (a0 - mean) * inv * g[t]       + bta[t];
  yr[t + 256] = (a1 - mean) * inv * g[t + 256] + bta[t + 256];
}

// ---------------------------------------------------------------------------
// host-side helpers
// ---------------------------------------------------------------------------
static inline void gemm(hipStream_t st, const float* A, const float* W,
                        const float* bias, const float* resid, float* C,
                        int M, int N, int K, int lda, int ldb, int ldc,
                        float alpha, int act,
                        int Z = 1, int Hdiv = 1, long sAb = 0, long sAh = 0,
                        long sBb = 0, long sBh = 0, long sCz = 0)
{
  dim3 grid(N / 128, M / 128, Z), block(256);
  gemm_bf16_kernel<<<grid, block, 0, st>>>(A, W, bias, resid, C, K, lda, ldb, ldc,
                                           sAb, sAh, sBb, sBh, sCz, Hdiv, alpha, act);
}

extern "C" void kernel_launch(void* const* d_in, const int* in_sizes, int n_in,
                              void* d_out, int out_size, void* d_ws, size_t ws_size,
                              hipStream_t stream)
{
  const int B = 4, L = 1024, SEQ = 1280, STR = 512, H = 8, FF = 2048;
  const int M = B * L;                    // 4096 rows
  const size_t NM = (size_t)M * STR;      // 2,097,152

  const float* seq = (const float*)d_in[0];
  const float* str = (const float*)d_in[1];
  const float* p_w = (const float*)d_in[2];
  const float* p_b = (const float*)d_in[3];
  const float* a1_qw = (const float*)d_in[4];  const float* a1_qb = (const float*)d_in[5];
  const float* a1_kw = (const float*)d_in[6];  const float* a1_kb = (const float*)d_in[7];
  const float* a1_vw = (const float*)d_in[8];  const float* a1_vb = (const float*)d_in[9];
  const float* a1_ow = (const float*)d_in[10]; const float* a1_ob = (const float*)d_in[11];
  const float* a2_qw = (const float*)d_in[12]; const float* a2_qb = (const float*)d_in[13];
  const float* a2_kw = (const float*)d_in[14]; const float* a2_kb = (const float*)d_in[15];
  const float* a2_vw = (const float*)d_in[16]; const float* a2_vb = (const float*)d_in[17];
  const float* a2_ow = (const float*)d_in[18]; const float* a2_ob = (const float*)d_in[19];
  const float* f1_w = (const float*)d_in[20]; const float* f1_b = (const float*)d_in[21];
  const float* f2_w = (const float*)d_in[22]; const float* f2_b = (const float*)d_in[23];
  const float* f3_w = (const float*)d_in[24]; const float* f3_b = (const float*)d_in[25];
  const float* f4_w = (const float*)d_in[26]; const float* f4_b = (const float*)d_in[27];
  const float* ln1_g = (const float*)d_in[28]; const float* ln1_b = (const float*)d_in[29];
  const float* ln2_g = (const float*)d_in[30]; const float* ln2_b = (const float*)d_in[31];
  const float* ln3_g = (const float*)d_in[32]; const float* ln3_b = (const float*)d_in[33];

  float* out = (float*)d_out;
  float* fuse_out = out;                         // (B,L,STR)
  float* str_out  = out + NM;                    // (B,L,STR)
  float* w1o = out + 2 * NM;                     // (B*H, L, L)
  float* w2o = out + 2 * NM + (size_t)16 * NM;   // (B*H, L, L)

  float* ws = (float*)d_ws;
  float* X   = ws;            // proj output / attn1 residual
  float* Qb  = ws + 1 * NM;
  float* Kb  = ws + 2 * NM;
  float* Vb  = ws + 3 * NM;
  float* CTX = ws + 4 * NM;
  float* Hln = ws + 5 * NM;
  float* F1  = ws + 6 * NM;
  float* F2  = ws + 7 * NM;
  float* F3  = ws + 8 * NM;
  float* HID = ws + 9 * NM;   // 4 * NM (4096 x 2048)
  float* RMAX = ws + 13 * NM; // B*H*L
  float* RSUM = RMAX + (size_t)B * H * L;

  const float qscale = 0.125f;  // 64^-0.5
  const long sBE = (long)L * STR;        // per-batch stride in (B,L,E)
  const long sLL = (long)L * L;          // per-(b,h) score stride
  const int  ZBH = B * H;

  // 1) x = seq @ p_w^T + p_b
  gemm(stream, seq, p_w, p_b, nullptr, X, M, STR, SEQ, SEQ, SEQ, STR, 1.f, 0);

  // ---- attention 1 (query = x, key/value = structure) ----
  gemm(stream, X,   a1_qw, a1_qb, nullptr, Qb, M, STR, STR, STR, STR, STR, qscale, 0);
  gemm(stream, str, a1_kw, a1_kb, nullptr, Kb, M, STR, STR, STR, STR, STR, 1.f, 0);
  gemm(stream, str, a1_vw, a1_vb, nullptr, Vb, M, STR, STR, STR, STR, STR, 1.f, 0);
  // raw scores -> w1 output region (batched over b,h)
  gemm(stream, Qb, Kb, nullptr, nullptr, w1o, L, L, 64, STR, STR, L, 1.f, 0,
       ZBH, H, sBE, 64, sBE, 64, sLL);
  softmax_stats_kernel<<<dim3(ZBH * L / 8), dim3(256), 0, stream>>>(w1o, RMAX, RSUM, L);
  attn_ctx_kernel<<<dim3(L / 128, ZBH), dim3(256), 0, stream>>>(w1o, RMAX, RSUM, Vb, CTX, L, STR, H);
  gemm(stream, CTX, a1_ow, a1_ob, X, F1, M, STR, STR, STR, STR, STR, 1.f, 0);

  // ---- FFN 1 ----
  ln_kernel<<<dim3(M), dim3(256), 0, stream>>>(F1, ln1_g, ln1_b, Hln);
  gemm(stream, Hln, f1_w, f1_b, nullptr, HID, M, FF, STR, STR, STR, FF, 1.f, 1 /*gelu*/);
  gemm(stream, HID, f2_w, f2_b, F1, F2, M, STR, FF, FF, FF, STR, 1.f, 0);

  // ---- attention 2 (self-attention on ln2(F2)) ----
  ln_kernel<<<dim3(M), dim3(256), 0, stream>>>(F2, ln2_g, ln2_b, Hln);
  gemm(stream, Hln, a2_qw, a2_qb, nullptr, Qb, M, STR, STR, STR, STR, STR, qscale, 0);
  gemm(stream, Hln, a2_kw, a2_kb, nullptr, Kb, M, STR, STR, STR, STR, STR, 1.f, 0);
  gemm(stream, Hln, a2_vw, a2_vb, nullptr, Vb, M, STR, STR, STR, STR, STR, 1.f, 0);
  gemm(stream, Qb, Kb, nullptr, nullptr, w2o, L, L, 64, STR, STR, L, 1.f, 0,
       ZBH, H, sBE, 64, sBE, 64, sLL);
  softmax_stats_kernel<<<dim3(ZBH * L / 8), dim3(256), 0, stream>>>(w2o, RMAX, RSUM, L);
  attn_ctx_kernel<<<dim3(L / 128, ZBH), dim3(256), 0, stream>>>(w2o, RMAX, RSUM, Vb, CTX, L, STR, H);
  gemm(stream, CTX, a2_ow, a2_ob, F2, F3, M, STR, STR, STR, STR, STR, 1.f, 0);

  // ---- FFN 2 -> final fuse straight into d_out ----
  ln_kernel<<<dim3(M), dim3(256), 0, stream>>>(F3, ln3_g, ln3_b, Hln);
  gemm(stream, Hln, f3_w, f3_b, nullptr, HID, M, FF, STR, STR, STR, FF, 1.f, 1 /*gelu*/);
  gemm(stream, HID, f4_w, f4_b, F3, fuse_out, M, STR, FF, FF, FF, STR, 1.f, 0);

  // pass-through output: structure_embed
  hipMemcpyAsync(str_out, str, NM * sizeof(float), hipMemcpyDeviceToDevice, stream);
}
